// MambaBlock_5360119186113
// MI455X (gfx1250) — compile-verified, matrix operand order checked by reading, not run on hardware
//
#include <hip/hip_runtime.h>
#include <hip/hip_bf16.h>

typedef __attribute__((ext_vector_type(16))) __bf16 v16bf;
typedef __attribute__((ext_vector_type(8)))  float  v8f;
typedef unsigned short ushort_t;
typedef unsigned int   uint_t;

#define BB   64
#define TT   2048
#define IND  256
#define ST   512
#define OD   256
#define KC3  3

// ---------------- helpers ----------------
__device__ __forceinline__ ushort_t f2bf(float f) {
    return __builtin_bit_cast(ushort_t, (__bf16)f);   // native v_cvt path
}
__device__ __forceinline__ float bf2f(ushort_t h) {
    return __builtin_bit_cast(float, ((uint_t)h) << 16);
}

union V16 {
    v16bf v;
    uint4 q[2];
    __device__ V16() {}
};

__device__ __forceinline__ v16bf frag_split(const ushort_t* p0, const ushort_t* p1) {
    V16 t;
    t.q[0] = *(const uint4*)p0;
    t.q[1] = *(const uint4*)p1;
    return t.v;
}
__device__ __forceinline__ v16bf frag_contig(const ushort_t* p) {
    V16 t;
    t.q[0] = ((const uint4*)p)[0];
    t.q[1] = ((const uint4*)p)[1];
    return t.v;
}
__device__ __forceinline__ v16bf frag_zero() {
    V16 t;
    t.q[0] = make_uint4(0u, 0u, 0u, 0u);
    t.q[1] = make_uint4(0u, 0u, 0u, 0u);
    return t.v;
}
__device__ __forceinline__ v8f wmma_bf16(v16bf a, v16bf b, v8f c) {
    return __builtin_amdgcn_wmma_f32_16x16x32_bf16(false, a, false, b, (short)0, c, false, false);
}

// ================= prep: x f32 -> bf16 (one pass; proj re-reads it 32x) =================
__global__ __launch_bounds__(256) void prep_x_kernel(const float* __restrict__ x,
                                                     ushort_t* __restrict__ xb) {
    size_t base = ((size_t)blockIdx.x * 256 + threadIdx.x) * 8;
    float4 a = ((const float4*)(x + base))[0];
    float4 b = ((const float4*)(x + base))[1];
    ushort4 o0, o1;
    o0.x = f2bf(a.x); o0.y = f2bf(a.y); o0.z = f2bf(a.z); o0.w = f2bf(a.w);
    o1.x = f2bf(b.x); o1.y = f2bf(b.y); o1.z = f2bf(b.z); o1.w = f2bf(b.w);
    ((ushort4*)(xb + base))[0] = o0;
    ((ushort4*)(xb + base))[1] = o1;
}

// ================= weight fragmentization =================
// B-operand fragment layout (internally consistent):
//   lane l: n = base_n + (l&15); kb = (l&16)?8:0
//   element e: k = kchunk*32 + kb + (e&7) + ((e&8)?16:0)
// Frag array flat index: ((tileId)*32 + lane)*16 + e, contiguous 32B per lane.

__global__ void prep_win_kernel(const float* __restrict__ W, ushort_t* __restrict__ dst) {
    int i = blockIdx.x * 256 + threadIdx.x;
    if (i >= ST * IND) return;
    int e = i & 15, l = (i >> 4) & 31, kc = (i >> 9) & 7, nt = i >> 12;
    int n = nt * 16 + (l & 15);
    int k = kc * 32 + ((l & 16) ? 8 : 0) + (e & 7) + ((e & 8) ? 16 : 0);
    dst[i] = f2bf(W[n * IND + k]);   // W_in is (ST, IN); GEMM-B = W_in^T[k][n]
}

__global__ void prep_mat_kernel(const float* __restrict__ src, ushort_t* __restrict__ dst,
                                int ncols, int total) {
    int i = blockIdx.x * 256 + threadIdx.x;
    if (i >= total) return;
    int e = i & 15, l = (i >> 4) & 31, kc = (i >> 9) & 15, nt = i >> 13;
    int n = nt * 16 + (l & 15);
    int k = kc * 32 + ((l & 16) ? 8 : 0) + (e & 7) + ((e & 8) ? 16 : 0);
    dst[i] = f2bf(src[k * ncols + n]);   // row-major (K, N)
}

__global__ void prep_conv_kernel(const float* __restrict__ cw, ushort_t* __restrict__ dst) {
    int i = blockIdx.x * 256 + threadIdx.x;
    if (i >= ST * ST * KC3) return;
    int e = i & 15, l = (i >> 4) & 31, cc = (i >> 9) & 15;
    int r = i >> 13;           // r = nt*3 + kk
    int kk = r % 3, nt = r / 3;
    int n = nt * 16 + (l & 15);
    int cin = cc * 32 + ((l & 16) ? 8 : 0) + (e & 7) + ((e & 8) ? 16 : 0);
    dst[i] = f2bf(cw[(n * ST + cin) * KC3 + kk]);   // conv_w (OUT, IN, K)
}

__global__ void zero_stats_kernel(float* __restrict__ p, int n) {
    for (int i = threadIdx.x; i < n; i += 256) p[i] = 0.f;
}

// ================= stage 1: x @ W_in^T + SiLU -> v (bf16) =================
__global__ __launch_bounds__(256) void proj_silu_kernel(
    const ushort_t* __restrict__ xb, const ushort_t* __restrict__ wf,
    const float* __restrict__ bias, ushort_t* __restrict__ v) {
    int tid = threadIdx.x, lane = tid & 31;
    int gw = blockIdx.x * 8 + (tid >> 5);        // global wave id
    int nt = gw & 31;                            // N tile (32 tiles of 16 -> 512)
    int rowBase = (gw >> 5) * 16;                // 8192 row tiles of 16
    int m = lane & 15;
    int kb = (lane & 16) ? 8 : 0;
    const ushort_t* xr = xb + (size_t)(rowBase + m) * IND;
    v8f acc = {0.f, 0.f, 0.f, 0.f, 0.f, 0.f, 0.f, 0.f};
#pragma unroll
    for (int kc = 0; kc < 8; ++kc) {
        int k0 = kc * 32 + kb;
        v16bf a = frag_split(xr + k0, xr + k0 + 16);
        v16bf b = frag_contig(wf + ((((size_t)nt * 8 + kc) * 32 + lane) << 4));
        acc = wmma_bf16(a, b, acc);
    }
    int n = nt * 16 + m;
    float bs = bias[n];
    int hi = (lane >> 4) << 3;
#pragma unroll
    for (int r = 0; r < 8; ++r) {
        float val = acc[r] + bs;
        val = val / (1.f + __expf(-val));        // SiLU
        v[(size_t)(rowBase + r + hi) * ST + n] = f2bf(val);
    }
}

// ================= stage 2: Conv1d(K=3) as 3 shifted GEMMs -> u (bf16, pre-BN) =================
__global__ __launch_bounds__(256) void conv_kernel(
    const ushort_t* __restrict__ v, const ushort_t* __restrict__ wf,
    const float* __restrict__ cbias, ushort_t* __restrict__ u) {
    int tid = threadIdx.x, lane = tid & 31;
    int gw = blockIdx.x * 8 + (tid >> 5);
    int nt = gw & 31;
    int rowBase = (gw >> 5) * 16;
    int m = lane & 15;
    int kb = (lane & 16) ? 8 : 0;
    int row = rowBase + m;
    int bt = row >> 11;          // batch (T=2048)
    int t = row & 2047;
    v8f acc = {0.f, 0.f, 0.f, 0.f, 0.f, 0.f, 0.f, 0.f};
#pragma unroll
    for (int kk = 0; kk < 3; ++kk) {
        int tp = t + kk - 1;
        bool ok = (unsigned)tp < (unsigned)TT;
        const ushort_t* vr = v + ((size_t)((bt << 11) | (tp & 2047))) * ST;
        for (int cc = 0; cc < 16; ++cc) {
            int k0 = cc * 32 + kb;
            v16bf a = ok ? frag_split(vr + k0, vr + k0 + 16) : frag_zero();
            const ushort_t* wp = wf + ((((size_t)(nt * 3 + kk) * 16 + cc) * 32 + lane) << 4);
            __builtin_prefetch(wp + 512, 0, 0);
            v16bf b = frag_contig(wp);
            acc = wmma_bf16(a, b, acc);
        }
    }
    int n = nt * 16 + m;
    float cb = cbias[n];
    int hi = (lane >> 4) << 3;
#pragma unroll
    for (int r = 0; r < 8; ++r)
        u[(size_t)(rowBase + r + hi) * ST + n] = f2bf(acc[r] + cb);
}

// ================= stage 3: BatchNorm stats / finalize / apply =================
__global__ __launch_bounds__(256) void bn_stats_kernel(
    const ushort_t* __restrict__ u, float* __restrict__ sum, float* __restrict__ sq) {
    __shared__ __align__(16) ushort_t tile[16 * ST];
    int tid = threadIdx.x;
    int sbase = blockIdx.x * 256;                // 256 samples per block
    float s0a = 0.f, s1a = 0.f, s0b = 0.f, s1b = 0.f;
    int ch0 = tid, ch1 = tid + 256;
    for (int tt = 0; tt < 16; ++tt) {
        __syncthreads();
#pragma unroll
        for (int j = 0; j < 2; ++j) {
            int idx = (tid + j * 256) * 16;      // covers 16*512 ushorts
            int srow = idx >> 9, col = idx & 511;
            const uint4* src = (const uint4*)(u + (size_t)(sbase + tt * 16 + srow) * ST + col);
            ((uint4*)(tile + idx))[0] = src[0];
            ((uint4*)(tile + idx))[1] = src[1];
        }
        __syncthreads();
#pragma unroll 4
        for (int r = 0; r < 16; ++r) {
            float a = bf2f(tile[r * ST + ch0]);
            float b = bf2f(tile[r * ST + ch1]);
            s0a += a; s1a += a * a;
            s0b += b; s1b += b * b;
        }
    }
    atomicAdd(&sum[ch0], s0a); atomicAdd(&sq[ch0], s1a);
    atomicAdd(&sum[ch1], s0b); atomicAdd(&sq[ch1], s1b);
}

__global__ void bn_finalize_kernel(const float* __restrict__ sum, const float* __restrict__ sq,
                                   const float* __restrict__ g, const float* __restrict__ b,
                                   float* __restrict__ scale, float* __restrict__ shift) {
    int i = blockIdx.x * 256 + threadIdx.x;
    if (i >= ST) return;
    const float invN = 1.f / (float)(BB * TT);
    float mu = sum[i] * invN;
    float var = sq[i] * invN - mu * mu;
    float sc = g[i] * rsqrtf(var + 1e-5f);
    scale[i] = sc;
    shift[i] = b[i] - mu * sc;
}

__global__ __launch_bounds__(256) void bn_apply_kernel(
    ushort_t* __restrict__ u, const float* __restrict__ scale, const float* __restrict__ shift) {
    size_t base = ((size_t)blockIdx.x * 256 + threadIdx.x) * 16;
    int ch0 = (int)(base & 511);
    uint4 q0 = ((uint4*)(u + base))[0];
    uint4 q1 = ((uint4*)(u + base))[1];
    ushort_t* p0 = (ushort_t*)&q0;
    ushort_t* p1 = (ushort_t*)&q1;
#pragma unroll
    for (int j = 0; j < 8; ++j) {
        p0[j] = f2bf(bf2f(p0[j]) * scale[ch0 + j] + shift[ch0 + j]);
        p1[j] = f2bf(bf2f(p1[j]) * scale[ch0 + 8 + j] + shift[ch0 + 8 + j]);
    }
    ((uint4*)(u + base))[0] = q0;
    ((uint4*)(u + base))[1] = q1;
}

// ================= stage 4: persistent sequential recurrence =================
// 4 workgroups (16 batches each) x 16 waves. State s (16x512 bf16) lives in LDS.
// Per step: async-stage v_t to LDS (ASYNCcnt path), s@A + v@Bm via 64 WMMAs/wave,
// LayerNorm via shfl-reduce + ds_add_f32 (double-buffered partials), y = s@C.
__global__ __launch_bounds__(512) void recurrence_kernel(
    const ushort_t* __restrict__ u, const ushort_t* __restrict__ Af,
    const ushort_t* __restrict__ Bf, const ushort_t* __restrict__ Cf,
    const float* __restrict__ lng, const float* __restrict__ lnb,
    float* __restrict__ out) {
    __shared__ __align__(16) ushort_t s_s[16 * ST];
    __shared__ __align__(16) ushort_t s_v[16 * ST];
    __shared__ float red[2][2][16];

    int tid = threadIdx.x, lane = tid & 31, w = tid >> 5;
    int b0 = blockIdx.x * 16;
    int m = lane & 15;
    int kb = (lane & 16) ? 8 : 0;
    int hi = (lane >> 4) << 3;

    // init: zero state + partial-sum buffers
    {
        int o = tid * 16;
        ((uint4*)(s_s + o))[0] = make_uint4(0u, 0u, 0u, 0u);
        ((uint4*)(s_s + o))[1] = make_uint4(0u, 0u, 0u, 0u);
        if (tid < 64) ((float*)red)[tid] = 0.f;
    }
    __syncthreads();

    int nt0 = 2 * w, nt1 = 2 * w + 1;
    int n0 = nt0 * 16 + m, n1 = nt1 * 16 + m;
    float lg0 = lng[n0], lb0 = lnb[n0];
    float lg1 = lng[n1], lb1 = lnb[n1];
    int nc = w * 16 + m;                        // C output column

    // per-lane LDS destination for async staging (flat->LDS uses addr[31:0])
    unsigned lds_v = (unsigned)(uintptr_t)(s_v + (w * ST + lane * 16));

    for (int t = 0; t < TT; ++t) {
        int p = t & 1;
        // ---- async stage v_t into LDS: GLOBAL_LOAD_ASYNC_TO_LDS_B128 x2 ----
        {
            const ushort_t* gptr = u + ((size_t)(b0 + w) * TT + t) * ST + lane * 16;
            asm volatile("global_load_async_to_lds_b128 %0, %1, off"
                         :: "v"(lds_v), "v"(gptr) : "memory");
            asm volatile("global_load_async_to_lds_b128 %0, %1, off offset:16"
                         :: "v"(lds_v), "v"(gptr) : "memory");
            asm volatile("s_wait_asynccnt 0x0" ::: "memory");
        }
        __syncthreads();

        // ---- acc = s@A + v@Bm for this wave's two 16-col tiles ----
        v8f acc0 = {0.f, 0.f, 0.f, 0.f, 0.f, 0.f, 0.f, 0.f};
        v8f acc1 = {0.f, 0.f, 0.f, 0.f, 0.f, 0.f, 0.f, 0.f};
        for (int kc = 0; kc < 16; ++kc) {
            int k0 = kc * 32 + kb;
            v16bf sa = frag_split(s_s + m * ST + k0, s_s + m * ST + k0 + 16);
            v16bf va = frag_split(s_v + m * ST + k0, s_v + m * ST + k0 + 16);
            const ushort_t* pa0 = Af + ((((size_t)nt0 * 16 + kc) * 32 + lane) << 4);
            const ushort_t* pb0 = Bf + ((((size_t)nt0 * 16 + kc) * 32 + lane) << 4);
            const ushort_t* pa1 = Af + ((((size_t)nt1 * 16 + kc) * 32 + lane) << 4);
            const ushort_t* pb1 = Bf + ((((size_t)nt1 * 16 + kc) * 32 + lane) << 4);
            __builtin_prefetch(pa0 + 1024, 0, 0);
            acc0 = wmma_bf16(sa, frag_contig(pa0), acc0);
            acc0 = wmma_bf16(va, frag_contig(pb0), acc0);
            acc1 = wmma_bf16(sa, frag_contig(pa1), acc1);
            acc1 = wmma_bf16(va, frag_contig(pb1), acc1);
        }

        // ---- LayerNorm partial sums: shfl over 16 N-lanes, ds atomics per row ----
#pragma unroll
        for (int r = 0; r < 8; ++r) {
            float xs = acc0[r] + acc1[r];
            float xq = acc0[r] * acc0[r] + acc1[r] * acc1[r];
#pragma unroll
            for (int off = 1; off < 16; off <<= 1) {
                xs += __shfl_xor(xs, off, 32);
                xq += __shfl_xor(xq, off, 32);
            }
            if ((lane & 15) == 0) {
                int mm = r + hi;
                atomicAdd(&red[p][0][mm], xs);
                atomicAdd(&red[p][1][mm], xq);
            }
        }
        __syncthreads();   // partials complete; all waves done reading old s

        // ---- normalize + write new s (bf16) to LDS; zero other-parity partials ----
#pragma unroll
        for (int r = 0; r < 8; ++r) {
            int mm = r + hi;
            float mu = red[p][0][mm] * (1.f / 512.f);
            float var = red[p][1][mm] * (1.f / 512.f) - mu * mu;
            float rs = rsqrtf(var + 1e-5f);
            s_s[mm * ST + n0] = f2bf((acc0[r] - mu) * rs * lg0 + lb0);
            s_s[mm * ST + n1] = f2bf((acc1[r] - mu) * rs * lg1 + lb1);
        }
        if (tid < 32) ((float*)(red[1 - p]))[tid] = 0.f;
        __syncthreads();   // new s visible

        // ---- y = s@C : each of 16 waves owns 16 of 256 output columns ----
        v8f yacc = {0.f, 0.f, 0.f, 0.f, 0.f, 0.f, 0.f, 0.f};
        for (int kc = 0; kc < 16; ++kc) {
            int k0 = kc * 32 + kb;
            v16bf sa = frag_split(s_s + m * ST + k0, s_s + m * ST + k0 + 16);
            v16bf cf = frag_contig(Cf + ((((size_t)w * 16 + kc) * 32 + lane) << 4));
            yacc = wmma_bf16(sa, cf, yacc);
        }
#pragma unroll
        for (int r = 0; r < 8; ++r) {
            int mm = r + hi;
            out[((size_t)(b0 + mm) * TT + t) * OD + nc] = yacc[r];
        }
    }
}

// ================= host launcher =================
extern "C" void kernel_launch(void* const* d_in, const int* in_sizes, int n_in,
                              void* d_out, int out_size, void* d_ws, size_t ws_size,
                              hipStream_t stream) {
    (void)in_sizes; (void)n_in; (void)out_size; (void)ws_size;
    const float* x      = (const float*)d_in[0];
    const float* W_in   = (const float*)d_in[1];
    const float* b_in   = (const float*)d_in[2];
    const float* conv_w = (const float*)d_in[3];
    const float* conv_b = (const float*)d_in[4];
    const float* bn_g   = (const float*)d_in[5];
    const float* bn_b   = (const float*)d_in[6];
    const float* ln_g   = (const float*)d_in[7];
    const float* ln_b   = (const float*)d_in[8];
    const float* A      = (const float*)d_in[9];
    const float* Bm     = (const float*)d_in[10];
    const float* C      = (const float*)d_in[11];
    float* out = (float*)d_out;

    char* ws = (char*)d_ws;
    const size_t X_BYTES  = (size_t)BB * TT * IND * 2;       // 64 MB
    const size_t V_BYTES  = (size_t)BB * TT * ST * 2;        // 128 MB
    const size_t WINF_B   = (size_t)ST * IND * 2;
    const size_t WCF_B    = (size_t)ST * ST * KC3 * 2;
    const size_t MAT_B    = (size_t)ST * ST * 2;
    const size_t CF_B     = (size_t)ST * OD * 2;

    size_t off = 0;
    ushort_t* x_bf   = (ushort_t*)(ws + off); off += X_BYTES;
    ushort_t* v_bf   = (ushort_t*)(ws + off); off += V_BYTES;
    ushort_t* u_bf   = (ushort_t*)(ws + off); off += V_BYTES;
    ushort_t* win_f  = (ushort_t*)(ws + off); off += WINF_B;
    ushort_t* wc_f   = (ushort_t*)(ws + off); off += WCF_B;
    ushort_t* a_f    = (ushort_t*)(ws + off); off += MAT_B;
    ushort_t* b_f    = (ushort_t*)(ws + off); off += MAT_B;
    ushort_t* c_f    = (ushort_t*)(ws + off); off += CF_B;
    float*    bn_sum = (float*)(ws + off);    off += ST * 4;
    float*    bn_sq  = (float*)(ws + off);    off += ST * 4;
    float*    bn_sc  = (float*)(ws + off);    off += ST * 4;
    float*    bn_sh  = (float*)(ws + off);    off += ST * 4;

    // prep: x conversion, weight fragmentization, stat zeroing
    zero_stats_kernel<<<1, 256, 0, stream>>>(bn_sum, ST * 2);   // sum+sq contiguous
    prep_x_kernel<<<(int)(((size_t)BB * TT * IND) / 8 / 256), 256, 0, stream>>>(x, x_bf);
    prep_win_kernel<<<(ST * IND) / 256, 256, 0, stream>>>(W_in, win_f);
    prep_conv_kernel<<<(ST * ST * KC3) / 256, 256, 0, stream>>>(conv_w, wc_f);
    prep_mat_kernel<<<(ST * ST) / 256, 256, 0, stream>>>(A, a_f, ST, ST * ST);
    prep_mat_kernel<<<(ST * ST) / 256, 256, 0, stream>>>(Bm, b_f, ST, ST * ST);
    prep_mat_kernel<<<(ST * OD) / 256, 256, 0, stream>>>(C, c_f, OD, ST * OD);

    // pipeline
    const int rowTiles = (BB * TT) / 16;                // 8192
    const int wavesPerGrid = rowTiles * 32;             // 262144 wave-tiles
    proj_silu_kernel<<<wavesPerGrid / 8, 256, 0, stream>>>(x_bf, win_f, b_in, v_bf);
    conv_kernel<<<wavesPerGrid / 8, 256, 0, stream>>>(v_bf, wc_f, conv_b, u_bf);
    bn_stats_kernel<<<(BB * TT) / 256, 256, 0, stream>>>(u_bf, bn_sum, bn_sq);
    bn_finalize_kernel<<<2, 256, 0, stream>>>(bn_sum, bn_sq, bn_g, bn_b, bn_sc, bn_sh);
    bn_apply_kernel<<<(int)(((size_t)BB * TT * ST) / 16 / 256), 256, 0, stream>>>(u_bf, bn_sc, bn_sh);
    recurrence_kernel<<<4, 512, 0, stream>>>(u_bf, a_f, b_f, c_f, ln_g, ln_b, out);
}
